// OutlookAttention2d_65644280152795
// MI455X (gfx1250) — matrix operand order repaired and checked by reading, not compiled, not run on hardware
//
#include <hip/hip_runtime.h>
#include <hip/hip_bf16.h>

typedef __attribute__((ext_vector_type(16))) _Float16 v16h;
typedef __attribute__((ext_vector_type(8)))  _Float16 v8h;
typedef __attribute__((ext_vector_type(8)))  float    v8f;

#define Bn   8
#define Cn   384
#define Hn   56
#define Wd   56
#define Ln   (Hn*Wd)      // 3136
#define NHEADS 6
#define KKn  9
#define APAD 64           // padded attn channels (54 -> 64)

// ---------------- fragment loaders (ISA 7.12.2 layouts, wave32) ----------------
// A-matrix 16x32 f16, row-major source (row stride = lda elements):
//   lane<16 : row=lane,    K = k0+0..7  and k0+16..23
//   lane>=16: row=lane-16, K = k0+8..15 and k0+24..31
__device__ __forceinline__ v16h load_a16x32(const _Float16* __restrict__ base,
                                            int row0, int lda, int k0, int lane) {
  const int lr = lane & 15, lh = lane >> 4;
  const _Float16* p = base + (size_t)(row0 + lr) * lda + k0 + lh * 8;
  v8h lo = *(const v8h*)p;
  v8h hi = *(const v8h*)(p + 16);
  return __builtin_shufflevector(lo, hi, 0,1,2,3,4,5,6,7,8,9,10,11,12,13,14,15);
}
// B-matrix 32x16 f16: lane holds column n=lane&15; lane<16: K 0..7,16..23; lane>=16: K 8..15,24..31.
// We feed it Bt = B^T stored row-major (row n contiguous in K), stride ldb.
__device__ __forceinline__ v16h load_bt32x16(const _Float16* __restrict__ base,
                                             int row0, int ldb, int k0, int lane) {
  const int lr = lane & 15, lh = lane >> 4;
  const _Float16* p = base + (size_t)(row0 + lr) * ldb + k0 + lh * 8;
  v8h lo = *(const v8h*)p;
  v8h hi = *(const v8h*)(p + 16);
  return __builtin_shufflevector(lo, hi, 0,1,2,3,4,5,6,7,8,9,10,11,12,13,14,15);
}

// ---------------- prep: transpose+convert x[B,C,L] f32 -> xT[B,L,C] f16 ----------------
__global__ void cvt_xT_kernel(const float* __restrict__ x, _Float16* __restrict__ xT) {
  int idx = blockIdx.x * blockDim.x + threadIdx.x;
  if (idx >= Bn * Ln * Cn) return;
  int c = idx % Cn;
  int r = idx / Cn;
  int l = r % Ln;
  int b = r / Ln;
  xT[idx] = (_Float16)x[((size_t)(b * Cn + c)) * Ln + l];
}

__global__ void cvt_w_kernel(const float* __restrict__ W, _Float16* __restrict__ Wh, int n) {
  int idx = blockIdx.x * blockDim.x + threadIdx.x;
  if (idx < n) Wh[idx] = (_Float16)W[idx];
}

// Wa [54,384] -> WaP [64,384] f16 (zero rows 54..63); ba -> baP[64] f32
__global__ void cvt_wa_kernel(const float* __restrict__ Wa, const float* __restrict__ ba,
                              _Float16* __restrict__ WaP, float* __restrict__ baP) {
  int idx = blockIdx.x * blockDim.x + threadIdx.x;
  if (idx < APAD * Cn) {
    int row = idx / Cn;
    WaP[idx] = (row < NHEADS * KKn) ? (_Float16)Wa[idx] : (_Float16)0.f;
  }
  if (idx < APAD) baP[idx] = (idx < NHEADS * KKn) ? ba[idx] : 0.f;
}

// ---------------- GEMM 1: v = x @ Wv^T + bv, pixel-major output vT[b,l,o] f16 ----------------
// D[m=l, n=o] = sum_c xT[l,c] * Wv[o,c]; wave tile 16(M) x 64(N)
__global__ __launch_bounds__(256) void gemm_v_kernel(const _Float16* __restrict__ xT,
                                                     const _Float16* __restrict__ Wv,
                                                     const float* __restrict__ bv,
                                                     _Float16* __restrict__ vT) {
  const int lane = threadIdx.x & 31;
  const int wid  = blockIdx.x * 8 + (threadIdx.x >> 5);
  const int MT = Ln / 16, NG = Cn / 64;           // 196, 6
  if (wid >= Bn * MT * NG) return;
  const int b  = wid / (MT * NG);
  const int rm = wid % (MT * NG);
  const int mt = rm / NG, ng = rm % NG;
  const int l0 = mt * 16, o0 = ng * 64;
  const int lr = lane & 15, lh = lane >> 4;
  const _Float16* Ab = xT + (size_t)b * Ln * Cn;

  v8f acc[4] = {};
  for (int k0 = 0; k0 < Cn; k0 += 32) {
    if (k0 + 32 < Cn)
      __builtin_prefetch((const void*)(Ab + (size_t)(l0 + lr) * Cn + k0 + 32), 0, 0);
    v16h af = load_a16x32(Ab, l0, Cn, k0, lane);
#pragma unroll
    for (int t = 0; t < 4; ++t) {
      v16h bf = load_bt32x16(Wv, o0 + t * 16, Cn, k0, lane);
      acc[t] = __builtin_amdgcn_wmma_f32_16x16x32_f16(false, af, false, bf,
                                                      (short)0, acc[t], false, false);
    }
  }
#pragma unroll
  for (int t = 0; t < 4; ++t) {
    const int oc = o0 + t * 16 + lr;
    const float bias = bv[oc];
#pragma unroll
    for (int r = 0; r < 8; ++r) {
      const int l = l0 + lh * 8 + r;
      vT[((size_t)(b * Ln + l)) * Cn + oc] = (_Float16)(acc[t][r] + bias);
    }
  }
}

// ---------------- GEMM 2: attn logits aT[b,l,0..63] f32 = x @ WaP^T + baP ----------------
__global__ __launch_bounds__(256) void gemm_a_kernel(const _Float16* __restrict__ xT,
                                                     const _Float16* __restrict__ WaP,
                                                     const float* __restrict__ baP,
                                                     float* __restrict__ aT) {
  const int lane = threadIdx.x & 31;
  const int wid  = blockIdx.x * 8 + (threadIdx.x >> 5);
  const int MT = Ln / 16;                          // 196
  if (wid >= Bn * MT) return;
  const int b  = wid / MT;
  const int mt = wid % MT;
  const int l0 = mt * 16;
  const int lr = lane & 15, lh = lane >> 4;
  const _Float16* Ab = xT + (size_t)b * Ln * Cn;

  v8f acc[4] = {};
  for (int k0 = 0; k0 < Cn; k0 += 32) {
    v16h af = load_a16x32(Ab, l0, Cn, k0, lane);
#pragma unroll
    for (int t = 0; t < 4; ++t) {
      v16h bf = load_bt32x16(WaP, t * 16, Cn, k0, lane);
      acc[t] = __builtin_amdgcn_wmma_f32_16x16x32_f16(false, af, false, bf,
                                                      (short)0, acc[t], false, false);
    }
  }
#pragma unroll
  for (int t = 0; t < 4; ++t) {
    const int oc = t * 16 + lr;
    const float bias = baP[oc];
#pragma unroll
    for (int r = 0; r < 8; ++r) {
      const int l = l0 + lh * 8 + r;
      aT[((size_t)(b * Ln + l)) * APAD + oc] = acc[t][r] + bias;
    }
  }
}

// ---------------- softmax over kk=9 + 3x3 window aggregation -> yT[b,l,c] f16 ----------------
__global__ __launch_bounds__(384) void attn_apply_kernel(const float* __restrict__ aT,
                                                         const _Float16* __restrict__ vT,
                                                         _Float16* __restrict__ yT) {
  __shared__ float sp[APAD];
  const int bl = blockIdx.x;                 // b*Ln + l
  const int b  = bl / Ln, l = bl % Ln;
  const int py = l / Wd, px = l % Wd;
  const int t  = threadIdx.x;

  if (t < APAD) sp[t] = aT[(size_t)bl * APAD + t];
  __syncthreads();
  if (t < NHEADS) {
    float mx = sp[t * KKn];
#pragma unroll
    for (int k = 1; k < KKn; ++k) mx = fmaxf(mx, sp[t * KKn + k]);
    float s = 0.f;
#pragma unroll
    for (int k = 0; k < KKn; ++k) {
      float e = __expf(sp[t * KKn + k] - mx);
      sp[t * KKn + k] = e;
      s += e;
    }
    float inv = 1.f / s;
#pragma unroll
    for (int k = 0; k < KKn; ++k) sp[t * KKn + k] *= inv;
  }
  __syncthreads();

  const int c = t;            // 0..383
  const int h = c >> 6;       // HD = 64
  float acc = 0.f;
#pragma unroll
  for (int k = 0; k < KKn; ++k) {
    const int yy = py + (k / 3) - 1;
    const int xx = px + (k % 3) - 1;
    if (yy >= 0 && yy < Hn && xx >= 0 && xx < Wd) {
      const int lk = yy * Wd + xx;
      acc += sp[h * KKn + k] * (float)vT[((size_t)(b * Ln + lk)) * Cn + c];
    }
  }
  yT[(size_t)bl * Cn + c] = (_Float16)acc;
}

// ---------------- GEMM 3: out[b,o,l] f32 = Wp @ y + bp ----------------
// D[m=o, n=l] = sum_c Wp[o,c] * yT[l,c]; wave tile 16(o) x 64(l)
__global__ __launch_bounds__(256) void gemm_p_kernel(const _Float16* __restrict__ yT,
                                                     const _Float16* __restrict__ Wp,
                                                     const float* __restrict__ bp,
                                                     float* __restrict__ out) {
  const int lane = threadIdx.x & 31;
  const int wid  = blockIdx.x * 8 + (threadIdx.x >> 5);
  const int OT = Cn / 16, LG = Ln / 64;            // 24, 49
  if (wid >= Bn * OT * LG) return;
  const int b  = wid / (OT * LG);
  const int rm = wid % (OT * LG);
  const int ot = rm / LG, lg = rm % LG;
  const int o0 = ot * 16, l0 = lg * 64;
  const int lr = lane & 15, lh = lane >> 4;
  const _Float16* Yb = yT + (size_t)b * Ln * Cn;

  v8f acc[4] = {};
  for (int k0 = 0; k0 < Cn; k0 += 32) {
    v16h af = load_a16x32(Wp, o0, Cn, k0, lane);
#pragma unroll
    for (int t = 0; t < 4; ++t) {
      v16h bf = load_bt32x16(Yb, l0 + t * 16, Cn, k0, lane);
      acc[t] = __builtin_amdgcn_wmma_f32_16x16x32_f16(false, af, false, bf,
                                                      (short)0, acc[t], false, false);
    }
  }
#pragma unroll
  for (int r = 0; r < 8; ++r) {
    const int o = o0 + lh * 8 + r;
    const float bias = bp[o];
#pragma unroll
    for (int t = 0; t < 4; ++t) {
      const int l = l0 + t * 16 + lr;
      out[((size_t)(b * Cn + o)) * Ln + l] = acc[t][r] + bias;
    }
  }
}

// ---------------- launch ----------------
extern "C" void kernel_launch(void* const* d_in, const int* in_sizes, int n_in,
                              void* d_out, int out_size, void* d_ws, size_t ws_size,
                              hipStream_t stream) {
  const float* x  = (const float*)d_in[0];
  const float* Wa = (const float*)d_in[1];
  const float* ba = (const float*)d_in[2];
  const float* Wv = (const float*)d_in[3];
  const float* bv = (const float*)d_in[4];
  const float* Wp = (const float*)d_in[5];
  const float* bp = (const float*)d_in[6];
  float* out = (float*)d_out;

  // workspace carve-up (all offsets 256B-aligned by construction)
  char* ws = (char*)d_ws;
  _Float16* xT  = (_Float16*)ws;                 ws += (size_t)Bn * Ln * Cn * sizeof(_Float16);
  _Float16* vT  = (_Float16*)ws;                 ws += (size_t)Bn * Ln * Cn * sizeof(_Float16);
  _Float16* yT  = (_Float16*)ws;                 ws += (size_t)Bn * Ln * Cn * sizeof(_Float16);
  float*    aT  = (float*)ws;                    ws += (size_t)Bn * Ln * APAD * sizeof(float);
  _Float16* WaP = (_Float16*)ws;                 ws += (size_t)APAD * Cn * sizeof(_Float16);
  _Float16* Wvh = (_Float16*)ws;                 ws += (size_t)Cn * Cn * sizeof(_Float16);
  _Float16* Wph = (_Float16*)ws;                 ws += (size_t)Cn * Cn * sizeof(_Float16);
  float*    baP = (float*)ws;                    ws += 256;

  // prep
  cvt_xT_kernel<<<(Bn * Ln * Cn + 255) / 256, 256, 0, stream>>>(x, xT);
  cvt_w_kernel<<<(Cn * Cn + 255) / 256, 256, 0, stream>>>(Wv, Wvh, Cn * Cn);
  cvt_w_kernel<<<(Cn * Cn + 255) / 256, 256, 0, stream>>>(Wp, Wph, Cn * Cn);
  cvt_wa_kernel<<<(APAD * Cn + 255) / 256, 256, 0, stream>>>(Wa, ba, WaP, baP);

  // attention logits (WMMA) then values (WMMA)
  {
    int waves = Bn * (Ln / 16);                       // 1568
    gemm_a_kernel<<<(waves + 7) / 8, 256, 0, stream>>>(xT, WaP, baP, aT);
  }
  {
    int waves = Bn * (Ln / 16) * (Cn / 64);           // 9408
    gemm_v_kernel<<<(waves + 7) / 8, 256, 0, stream>>>(xT, Wvh, bv, vT);
  }

  // softmax + 3x3 outlook aggregation
  attn_apply_kernel<<<Bn * Ln, 384, 0, stream>>>(aT, vT, yT);

  // projection (WMMA) -> f32 NCHW output
  {
    int waves = Bn * (Cn / 16) * (Ln / 64);           // 9408
    gemm_p_kernel<<<(waves + 7) / 8, 256, 0, stream>>>(yT, Wph, bp, out);
  }
}
